// SioConv_33294586478859
// MI455X (gfx1250) — matrix operand order repaired and checked by reading, not compiled, not run on hardware
//
#include <hip/hip_runtime.h>
#include <math.h>

// SioConv: raw = x @ W_a^T  (B*L x 2D, K=D), then per (b,d) complex suffix scan:
//   a_l = (re+i im) * rsqrt(|.|^2) * exp(-|.|^2)
//   s[i] = a_i * (xhat_i + s[i+1]),  h[i] = s[i] + xhat_{i-1},  h[L-1] += x[L-1]
// B=2, L=256, D=256.

typedef __attribute__((ext_vector_type(16))) _Float16 v16h;
typedef __attribute__((ext_vector_type(8)))  float    v8f;

#define BB 2
#define LL 256
#define DD 256

__device__ __forceinline__ void splitf(float f, _Float16& hi, _Float16& lo) {
    hi = (_Float16)f;
    lo = (_Float16)(f - (float)hi);
}

// GEMM: per block: batch b, 32 d-values (64 W columns), all 256 rows (l).
// 256 threads = 8 waves; wave w owns M-tiles {2w,2w+1} x N-tiles {0..3}.
// Split-f16 WMMA: acc += Ahi*Bhi + Ahi*Blo + Alo*Bhi  (f32 accumulate).
__global__ __launch_bounds__(256)
void sioconv_gemm_wmma(const float* __restrict__ x,
                       const float* __restrict__ Wa,
                       float* __restrict__ raw) {
    __shared__ _Float16 lxh[LL][32];   // x slice hi: [l][k]
    __shared__ _Float16 lxl[LL][32];   // x slice lo
    __shared__ _Float16 lwh[64][32];   // W slice hi: [col][k]
    __shared__ _Float16 lwl[64][32];   // W slice lo

    const int b  = blockIdx.y;
    const int d0 = blockIdx.x * 32;          // covers W rows [2*d0, 2*d0+64)
    const int t  = threadIdx.x;
    const int wv = t >> 5;                   // wave 0..7
    const int ln = t & 31;                   // lane in wave32
    const int r  = ln & 15;
    const int hh = ln >> 4;                  // half-wave selector

    v8f acc[2][4] = {};

    for (int kt = 0; kt < DD / 32; ++kt) {
        const int k0 = kt * 32;
        __syncthreads();
        // stage x[b, l=t, k0..k0+31] -> LDS (split hi/lo)
        {
            const float* src = x + ((size_t)b * LL + t) * DD + k0;
            #pragma unroll
            for (int k = 0; k < 32; ++k)
                splitf(src[k], lxh[t][k], lxl[t][k]);
        }
        // stage Wa[2*d0 + c, k0..k0+31] -> LDS, c = t>>2, 8 floats each
        {
            const int c  = t >> 2;
            const int kk = (t & 3) * 8;
            const float* src = Wa + (size_t)(2 * d0 + c) * DD + k0 + kk;
            #pragma unroll
            for (int k = 0; k < 8; ++k)
                splitf(src[k], lwh[c][kk + k], lwl[c][kk + k]);
        }
        __syncthreads();

        // Build fragments per documented VGPR layouts.
        // A 16x32 f16: lane r (0-15) row M=r; halves 0..7 -> K=hh*8+j, 8..15 -> K=16+hh*8+j
        v16h ah[2], al[2], bh[4], bl[4];
        #pragma unroll
        for (int mi = 0; mi < 2; ++mi) {
            const int row = (wv * 2 + mi) * 16 + r;
            #pragma unroll
            for (int j = 0; j < 8; ++j) {
                ah[mi][j]     = lxh[row][hh * 8 + j];
                ah[mi][8 + j] = lxh[row][16 + hh * 8 + j];
                al[mi][j]     = lxl[row][hh * 8 + j];
                al[mi][8 + j] = lxl[row][16 + hh * 8 + j];
            }
        }
        // B 32x16 f16: lane r = col N; halves 0..7 -> K=hh*8+e, 8..15 -> K=16+hh*8+(e-8)
        #pragma unroll
        for (int ni = 0; ni < 4; ++ni) {
            const int c = ni * 16 + r;
            #pragma unroll
            for (int e = 0; e < 16; ++e) {
                bh[ni][e] = lwh[c][(e >> 3) * 16 + hh * 8 + (e & 7)];
                bl[ni][e] = lwl[c][(e >> 3) * 16 + hh * 8 + (e & 7)];
            }
        }

        #pragma unroll
        for (int mi = 0; mi < 2; ++mi)
            #pragma unroll
            for (int ni = 0; ni < 4; ++ni) {
                v8f a = acc[mi][ni];
                a = __builtin_amdgcn_wmma_f32_16x16x32_f16(false, al[mi], false, bh[ni], (short)0, a, false, false);
                a = __builtin_amdgcn_wmma_f32_16x16x32_f16(false, ah[mi], false, bl[ni], (short)0, a, false, false);
                a = __builtin_amdgcn_wmma_f32_16x16x32_f16(false, ah[mi], false, bh[ni], (short)0, a, false, false);
                acc[mi][ni] = a;
            }
    }

    // C/D layout: lane -> N = r, M = hh*8 + v
    #pragma unroll
    for (int mi = 0; mi < 2; ++mi)
        #pragma unroll
        for (int ni = 0; ni < 4; ++ni)
            #pragma unroll
            for (int v = 0; v < 8; ++v) {
                const int l   = (wv * 2 + mi) * 16 + hh * 8 + v;
                const int col = ni * 16 + r;
                raw[((size_t)b * LL + l) * (2 * DD) + (size_t)(2 * d0) + col] = acc[mi][ni][v];
            }
}

// Suffix scan: one lane per (b, d). 512 lanes total, 256 serial complex steps.
__global__ __launch_bounds__(256)
void sioconv_scan(const float* __restrict__ x,
                  const float* __restrict__ raw,
                  const float* __restrict__ h0r_,
                  const float* __restrict__ h0i_,
                  float* __restrict__ out) {
    const int b = blockIdx.x;
    const int d = threadIdx.x;
    const float h0r = h0r_[d];
    const float h0i = h0i_[d];
    float sr = 0.f, si = 0.f;
    for (int i = LL - 1; i >= 0; --i) {
        const size_t rbase = ((size_t)b * LL + i) * (2 * DD) + 2 * d;
        const float re = raw[rbase];
        const float im = raw[rbase + 1];
        const float m2 = re * re + im * im;
        const float sc = rsqrtf(m2) * expf(-m2);
        const float ar = re * sc, ai = im * sc;
        float xr, xi;
        if (i == 0) { xr = h0r; xi = h0i; }
        else        { xr = x[((size_t)b * LL + (i - 1)) * DD + d]; xi = 0.f; }
        const float tr = xr + sr, ti = xi + si;
        sr = ar * tr - ai * ti;
        si = ar * ti + ai * tr;
        float add = 0.f;
        if (i >= 2)      add = x[((size_t)b * LL + (i - 2)) * DD + d];
        else if (i == 1) add = h0r;
        float o = sr + add;
        if (i == LL - 1) o += x[((size_t)b * LL + (LL - 1)) * DD + d];
        out[((size_t)b * LL + i) * DD + d] = o;
    }
}

extern "C" void kernel_launch(void* const* d_in, const int* in_sizes, int n_in,
                              void* d_out, int out_size, void* d_ws, size_t ws_size,
                              hipStream_t stream) {
    const float* x   = (const float*)d_in[0];   // (B, L, D) f32
    const float* Wa  = (const float*)d_in[1];   // (2D, D)   f32
    const float* h0r = (const float*)d_in[2];   // (D,)      f32
    const float* h0i = (const float*)d_in[3];   // (D,)      f32
    float* out = (float*)d_out;                 // (B, L, D) f32
    float* raw = (float*)d_ws;                  // B*L*2D floats = 1 MB scratch

    sioconv_gemm_wmma<<<dim3(DD / 32, BB), 256, 0, stream>>>(x, Wa, raw);
    sioconv_scan<<<dim3(BB), 256, 0, stream>>>(x, raw, h0r, h0i, out);
}
